// Lstm_75574244540721
// MI455X (gfx1250) — matrix operand order repaired and checked by reading, not compiled, not run on hardware
//
#include <hip/hip_runtime.h>

// ---------------------------------------------------------------------------
// Bidirectional packed LSTM for MI455X (gfx1250, wave32, WMMA).
// 4 "directions" = {fwd c0, fwd c1, bwd c0, bwd c1}. Per time step, each
// (dir, h-tile) workgroup computes gates = x_t*Wih^T + h*Whh^T + bias via
// v_wmma_f32_16x16x32_f16, applies the LSTM nonlinearity, updates c (f32,
// global ws) and h (f16, double-buffered global ws), and scatters h into the
// packed output (bwd writes to the time-reversed row).
// ---------------------------------------------------------------------------

typedef __attribute__((ext_vector_type(16))) _Float16 v16h;
typedef __attribute__((ext_vector_type(8)))  _Float16 h8;
typedef __attribute__((ext_vector_type(8)))  float    v8f;

#define HDIM 256
#define IDIM 256
#define BMAX 128
#define CCH  2
#define GDIM 1024          // 4*H gate dimension
#define NDIR 4

// Workspace layout (bytes). Total ~5.27 MB (output buffer is ~101 MB, so the
// harness workspace is expected to cover this comfortably).
#define WS_OFF    ((size_t)0)          // int[257]  packed-seq offsets
#define WS_LEN    ((size_t)2048)       // int[128]  per-row lengths
#define WS_BIAS   ((size_t)4096)       // float[4*1024]  bias_ih+bias_hh per dir
#define WS_C      ((size_t)20480)      // float[4*128*256]  cell state
#define WS_H0     ((size_t)544768)     // f16  [4*128*256]  hidden buf 0
#define WS_H1     ((size_t)806912)     // f16  [4*128*256]  hidden buf 1
#define WS_WIH    ((size_t)1069056)    // f16  [4*1024*256] W_ih per dir
#define WS_WHH    ((size_t)3166208)    // f16  [4*1024*256] W_hh per dir
// end: 5263360 bytes

__device__ __forceinline__ float sigmf(float x) {
  return 1.0f / (1.0f + __expf(-x));
}
__device__ __forceinline__ float tanhfast(float x) {
  float xx = fminf(fmaxf(x, -15.0f), 15.0f);
  float e  = __expf(2.0f * xx);
  return (e - 1.0f) / (e + 1.0f);
}
__device__ __forceinline__ v16h mk16(h8 lo, h8 hi) {
  v16h r;
#pragma unroll
  for (int e = 0; e < 8; ++e) { r[e] = lo[e]; r[e + 8] = hi[e]; }
  return r;
}

// ---------------------------------------------------------------------------
// Setup: offsets = exclusive cumsum(batch_sizes); lengths[b] = #t with b<bs[t]
// ---------------------------------------------------------------------------
__global__ void bilstm_setup(const int* __restrict__ bs, int T,
                             int* __restrict__ offs, int* __restrict__ lens) {
  const int tid = threadIdx.x;
  if (tid == 0) {
    int acc = 0;
    for (int t = 0; t < T; ++t) { offs[t] = acc; acc += bs[t]; }
    offs[T] = acc;
  }
  if (tid < BMAX) {
    int cnt = 0;
    for (int t = 0; t < T; ++t) cnt += (tid < bs[t]) ? 1 : 0;
    lens[tid] = cnt;
  }
}

// ---------------------------------------------------------------------------
// Init: f32 -> f16 weight conversion into per-direction layout, fused bias,
// zero h0/c.  idx enumerates [dir][j][k], k fastest (1,048,576 threads).
// ---------------------------------------------------------------------------
__global__ __launch_bounds__(256) void bilstm_init(
    const float* __restrict__ wih,  const float* __restrict__ whh,
    const float* __restrict__ wihr, const float* __restrict__ whhr,
    const float* __restrict__ bih,  const float* __restrict__ bhh,
    const float* __restrict__ bihr, const float* __restrict__ bhhr,
    unsigned short* __restrict__ wih_o, unsigned short* __restrict__ whh_o,
    float* __restrict__ bias_o, unsigned short* __restrict__ h0_o,
    float* __restrict__ c0) {
  const int idx = blockIdx.x * blockDim.x + threadIdx.x;   // < 4*1024*256
  _Float16* WI = reinterpret_cast<_Float16*>(wih_o);
  _Float16* WH = reinterpret_cast<_Float16*>(whh_o);
  const int dir = idx >> 18;           // 262144 elems per dir
  const int rem = idx & 0x3FFFF;       // j*256 + k
  const float vi = (dir < 2) ? wih[(size_t)dir * 262144 + rem]
                             : wihr[(size_t)(dir - 2) * 262144 + rem];
  const float vh = (dir < 2) ? whh[(size_t)dir * 262144 + rem]
                             : whhr[(size_t)(dir - 2) * 262144 + rem];
  WI[idx] = (_Float16)vi;
  WH[idx] = (_Float16)vh;
  if (idx < NDIR * GDIM) {
    const int d = idx >> 10, j = idx & 1023;
    bias_o[idx] = (d < 2) ? (bih[d * GDIM + j] + bhh[d * GDIM + j])
                          : (bihr[(d - 2) * GDIM + j] + bhhr[(d - 2) * GDIM + j]);
  }
  if (idx < NDIR * BMAX * HDIM) {
    reinterpret_cast<_Float16*>(h0_o)[idx] = (_Float16)0.0f;
    c0[idx] = 0.0f;
  }
}

// ---------------------------------------------------------------------------
// One recurrence step. grid = (4 dirs, 16 h-tiles), block = 256 (8 waves).
// Wave w owns M-tile w (batch rows 16w..16w+15) and the 4 gate tiles (i,f,g,o)
// for h columns [16*ht, 16*ht+16).  K = 512 = [x (256) | h (256)], staged in
// LDS in 4 chunks of 128.
// ---------------------------------------------------------------------------
__global__ __launch_bounds__(256) void bilstm_step(
    int t,
    const float* __restrict__ data,
    const unsigned short* __restrict__ wih_u,
    const unsigned short* __restrict__ whh_u,
    const float* __restrict__ bias,
    const int* __restrict__ offs, const int* __restrict__ lens,
    const int* __restrict__ bs,
    const unsigned short* __restrict__ hin_u, unsigned short* __restrict__ hout_u,
    float* __restrict__ cws, float* __restrict__ out) {
  const _Float16* Wih = reinterpret_cast<const _Float16*>(wih_u);
  const _Float16* Whh = reinterpret_cast<const _Float16*>(whh_u);
  const _Float16* hin = reinterpret_cast<const _Float16*>(hin_u);
  _Float16* hout = reinterpret_cast<_Float16*>(hout_u);

  const int dir = blockIdx.x;          // 0..3
  const int ht  = blockIdx.y;          // 0..15
  const int cch = dir & 1;
  const bool rev = (dir >= 2);
  const int tid  = threadIdx.x;
  const int wave = tid >> 5;
  const int lane = tid & 31;

  // Row stride 136 f16 = 272 B = 17*16 B -> 16-B aligned b128 LDS accesses.
  __shared__ _Float16 Alds[128][136];  // A chunk: 128 batch rows x 128 K
  __shared__ _Float16 Blds[64][136];   // B chunk: 4 gates x 16 cols x 128 K

  const int bs_t  = bs[t];
  const int off_t = offs[t];

  const v8f vzero = {0.f, 0.f, 0.f, 0.f, 0.f, 0.f, 0.f, 0.f};
  v8f acc[4];
  acc[0] = vzero; acc[1] = vzero; acc[2] = vzero; acc[3] = vzero;

  // A-staging assignment: 2 threads per batch row, 64 contiguous K each.
  const int bA    = tid >> 1;
  const int halfA = tid & 1;
  int  validA;
  long srcnA = 0;
  if (!rev) { validA = (bA < bs_t); srcnA = off_t + bA; }
  else {
    const int lb = lens[bA];
    validA = (t < lb);
    srcnA  = validA ? (long)(offs[lb - 1 - t] + bA) : 0;
  }

  // B-staging assignment: 4 threads per weight row, 32 contiguous K each.
  const int rB = tid >> 2;                       // 0..63 = gate*16 + col
  const int qB = tid & 3;
  const int jB = (rB >> 4) * HDIM + ht * 16 + (rB & 15);

  // Fragment addressing (constant across chunks).
  const int mrow = wave * 16 + (lane & 15);
  const int hi   = lane >> 4;                    // lane half selects K half
  const int coln = lane & 15;

  for (int kc = 0; kc < 4; ++kc) {
    const int kbase = kc * 128;
    __syncthreads();                             // protect previous chunk reads
    // ---- stage A (x for kc<2, h_prev for kc>=2), converted to f16
    {
      _Float16* dst = &Alds[bA][halfA * 64];
      if (kc < 2) {
        const float* src =
            data + ((size_t)srcnA * CCH + cch) * IDIM + kbase + halfA * 64;
        if (validA) {
#pragma unroll
          for (int e = 0; e < 64; e += 4) {
            float4 v = *reinterpret_cast<const float4*>(src + e);
            dst[e + 0] = (_Float16)v.x; dst[e + 1] = (_Float16)v.y;
            dst[e + 2] = (_Float16)v.z; dst[e + 3] = (_Float16)v.w;
          }
        } else {
#pragma unroll
          for (int e = 0; e < 64; ++e) dst[e] = (_Float16)0.0f;
        }
      } else {
        const _Float16* src =
            hin + ((size_t)dir * BMAX + bA) * HDIM + (kbase - 256) + halfA * 64;
#pragma unroll
        for (int e = 0; e < 64; e += 8)
          *reinterpret_cast<h8*>(dst + e) =
              *reinterpret_cast<const h8*>(src + e);
      }
    }
    // ---- stage B (Wih for kc<2, Whh for kc>=2)
    {
      const _Float16* Wsrc =
          (kc < 2) ? (Wih + ((size_t)dir * GDIM + jB) * IDIM + kbase + qB * 32)
                   : (Whh + ((size_t)dir * GDIM + jB) * HDIM + (kbase - 256) + qB * 32);
      if (kc == 1)  // warm the recurrent-weight rows (global_prefetch_b8)
        __builtin_prefetch(Whh + ((size_t)dir * GDIM + jB) * HDIM + qB * 32, 0, 0);
      _Float16* dst = &Blds[rB][qB * 32];
#pragma unroll
      for (int e = 0; e < 32; e += 8)
        *reinterpret_cast<h8*>(dst + e) =
            *reinterpret_cast<const h8*>(Wsrc + e);
    }
    __syncthreads();
    // ---- 4 WMMA k-steps of 32 over this chunk.
    // Batch the A fragment and ALL four gate B fragments into distinct
    // registers before issuing the WMMAs, so the scheduler can issue the
    // ds_load_b128s back-to-back and use partial s_wait_dscnt thresholds
    // instead of a full wait before every WMMA.
#pragma unroll
    for (int ks = 0; ks < 4; ++ks) {
      // A (16x32 f16): lanes0-15 hold K[0..7],[16..23]; lanes16-31 +8
      const int s0 = ks * 32 + hi * 8;
      h8 a0 = *reinterpret_cast<const h8*>(&Alds[mrow][s0]);
      h8 a1 = *reinterpret_cast<const h8*>(&Alds[mrow][s0 + 16]);
      // B (32x16 f16): lanes0-15 col n K[0..7],[16..23]; lanes16-31 +8
      const int bk = ks * 32 + hi * 8;
      h8 b0[4], b1[4];
#pragma unroll
      for (int g = 0; g < 4; ++g) {
        b0[g] = *reinterpret_cast<const h8*>(&Blds[g * 16 + coln][bk]);
        b1[g] = *reinterpret_cast<const h8*>(&Blds[g * 16 + coln][bk + 16]);
      }
      const v16h av = mk16(a0, a1);
#pragma unroll
      for (int g = 0; g < 4; ++g) {
        const v16h bv = mk16(b0[g], b1[g]);
        acc[g] = __builtin_amdgcn_wmma_f32_16x16x32_f16(
            false, av, false, bv, (short)0, acc[g], false, false);
      }
    }
  }

  // ---- epilogue: bias + LSTM nonlinearity + state update + packed scatter
  const int hidx = ht * 16 + coln;
  const float bi = bias[dir * GDIM + 0 * HDIM + hidx];
  const float bff= bias[dir * GDIM + 1 * HDIM + hidx];
  const float bg = bias[dir * GDIM + 2 * HDIM + hidx];
  const float bo = bias[dir * GDIM + 3 * HDIM + hidx];
  const int mhi  = hi * 8;                       // C/D layout: lanes16-31 -> M+8

#pragma unroll
  for (int r = 0; r < 8; ++r) {
    const int b = wave * 16 + mhi + r;           // batch row of this element
    int valid, outrow;
    if (!rev) { valid = (b < bs_t); outrow = off_t + b; }
    else {
      const int lb = lens[b];
      valid  = (t < lb);
      outrow = valid ? (offs[lb - 1 - t] + b) : 0;
    }
    const float gi = acc[0][r] + bi;
    const float gf = acc[1][r] + bff;
    const float gg = acc[2][r] + bg;
    const float go = acc[3][r] + bo;
    const size_t sidx = ((size_t)dir * BMAX + b) * HDIM + hidx;
    const float cprev = cws[sidx];
    const float cn = sigmf(gi) * tanhfast(gg) + sigmf(gf) * cprev;
    const float hn = sigmf(go) * tanhfast(cn);
    const _Float16 hprev = hin[sidx];
    if (valid) {
      cws[sidx]  = cn;
      hout[sidx] = (_Float16)hn;
      out[(size_t)outrow * (CCH * 2 * HDIM) + cch * (2 * HDIM) +
          (rev ? HDIM : 0) + hidx] = hn;
    } else {
      hout[sidx] = hprev;                        // carry frozen state forward
    }
  }
}

// ---------------------------------------------------------------------------
extern "C" void kernel_launch(void* const* d_in, const int* in_sizes, int n_in,
                              void* d_out, int out_size, void* d_ws,
                              size_t ws_size, hipStream_t stream) {
  (void)n_in; (void)out_size; (void)ws_size;
  const float* data = (const float*)d_in[0];
  const float* wih  = (const float*)d_in[1];
  const float* whh  = (const float*)d_in[2];
  const float* bih  = (const float*)d_in[3];
  const float* bhh  = (const float*)d_in[4];
  const float* wihr = (const float*)d_in[5];
  const float* whhr = (const float*)d_in[6];
  const float* bihr = (const float*)d_in[7];
  const float* bhhr = (const float*)d_in[8];
  const int*   bs   = (const int*)d_in[9];
  const int T = in_sizes[9];                     // 256

  char* ws = (char*)d_ws;
  int*            offs   = (int*)(ws + WS_OFF);
  int*            lens   = (int*)(ws + WS_LEN);
  float*          biasws = (float*)(ws + WS_BIAS);
  float*          cws    = (float*)(ws + WS_C);
  unsigned short* h0     = (unsigned short*)(ws + WS_H0);
  unsigned short* h1     = (unsigned short*)(ws + WS_H1);
  unsigned short* Wihw   = (unsigned short*)(ws + WS_WIH);
  unsigned short* Whhw   = (unsigned short*)(ws + WS_WHH);
  float* out = (float*)d_out;

  bilstm_setup<<<1, 256, 0, stream>>>(bs, T, offs, lens);
  bilstm_init<<<4096, 256, 0, stream>>>(wih, whh, wihr, whhr, bih, bhh, bihr,
                                        bhhr, Wihw, Whhw, biasws, h0, cws);
  for (int t = 0; t < T; ++t) {
    unsigned short* hin  = (t & 1) ? h1 : h0;
    unsigned short* hout = (t & 1) ? h0 : h1;
    bilstm_step<<<dim3(NDIR, 16), 256, 0, stream>>>(
        t, data, Wihw, Whhw, biasws, offs, lens, bs, hin, hout, cws, out);
  }
}
